// Decoder_25761213841585
// MI455X (gfx1250) — matrix operand order repaired and checked by reading, not compiled, not run on hardware
//
#include <hip/hip_runtime.h>
#include <hip/hip_bf16.h>
#include <math.h>

// ---------------- problem constants ----------------
#define NWG   32
#define NTHR  256
#define GSZ   (NWG * NTHR)
#define Bsz   32
#define Tlen  256
#define Hdim  512

typedef __bf16 v16bf __attribute__((ext_vector_type(16)));
typedef float  v8f   __attribute__((ext_vector_type(8)));
typedef unsigned uint4a __attribute__((ext_vector_type(4)));

union VecBF { uint4a u[2]; v16bf v; };

// ---------------- workspace layout (bytes) ----------------
constexpr size_t OFF_BAR   = 0;                                   // 256B barrier state
constexpr size_t OFF_WIH   = 256;                                 // [2048,2048] bf16
constexpr size_t OFF_WHH   = OFF_WIH   + (size_t)2048*2048*2;     // [2048,512]  bf16
constexpr size_t OFF_WINS  = OFF_WHH   + (size_t)2048*512*2;      // [512,512]   bf16
constexpr size_t OFF_WOUTS = OFF_WINS  + (size_t)512*512*2;       // [512,1024]  bf16
constexpr size_t OFF_WINV  = OFF_WOUTS + (size_t)512*1024*2;      // [512,512]   bf16
constexpr size_t OFF_WOUTV = OFF_WINV  + (size_t)512*512*2;       // [512,1024]  bf16
constexpr size_t OFF_WAL   = OFF_WOUTV + (size_t)512*1024*2;      // [512,1536]  bf16
constexpr size_t OFF_BUF   = OFF_WAL   + (size_t)512*1536*2;      // h history [B,T,512] bf16
constexpr size_t OFF_XBF   = OFF_BUF   + (size_t)Bsz*Tlen*512*2;  // x step [B,2048] bf16
constexpr size_t OFF_CAT1  = OFF_XBF   + (size_t)Bsz*2048*2;      // [cvec1;h]  [B,1024] bf16
constexpr size_t OFF_CAT2  = OFF_CAT1  + (size_t)Bsz*1024*2;      // [cvec2;so] [B,1024] bf16
constexpr size_t OFF_CAT3  = OFF_CAT2  + (size_t)Bsz*1024*2;      // [h;so;vo]  [B,1536] bf16
constexpr size_t OFF_GATES = OFF_CAT3  + (size_t)Bsz*1536*2;      // [B,2048] f32
constexpr size_t OFF_C     = OFF_GATES + (size_t)Bsz*2048*4;      // c state [B,512] f32
constexpr size_t OFF_QP    = OFF_C     + (size_t)Bsz*512*4;       // [B,512] f32
constexpr size_t OFF_QP2   = OFF_QP    + (size_t)Bsz*512*4;       // [B,512] f32

// ---------------- helpers ----------------
__device__ __forceinline__ float sigf(float x) { return 1.0f / (1.0f + expf(-x)); }

// Accumulate a 16x16 output tile of A[M,K] (row-major, bf16) * W[N,K]^T (row-major, bf16)
// using v_wmma_f32_16x16x32_bf16. K must be a multiple of 32.
__device__ __forceinline__ v8f wmma_accum(v8f acc,
                                          const __bf16* __restrict__ A, int lda,
                                          const __bf16* __restrict__ W, int ldw,
                                          int m0, int n0, int K, int lane) {
  const int lm = lane & 15;
  const int hi = lane >> 4;                 // 0 for lanes 0-15, 1 for lanes 16-31
  // A 16x32 bf16 layout: lane m, low half K 0-7 & 16-23, high half K 8-15 & 24-31
  const __bf16* pa = A + (size_t)(m0 + lm) * lda + (hi << 3);
  // B 32x16 bf16 layout (W^T tile): lane n, low half K 0-7 & 16-23, high half K 8-15 & 24-31
  const __bf16* pb = W + (size_t)(n0 + lm) * ldw + (hi << 3);
  for (int k = 0; k < K; k += 32) {
    VecBF a, b;
    a.u[0] = *(const uint4a*)(pa);
    a.u[1] = *(const uint4a*)(pa + 16);
    b.u[0] = *(const uint4a*)(pb);
    b.u[1] = *(const uint4a*)(pb + 16);
    acc = __builtin_amdgcn_wmma_f32_16x16x32_bf16(false, a.v, false, b.v,
                                                  (short)0, acc, false, false);
    pa += 32;
    pb += 32;
  }
  return acc;
}

// C/D 16x16 f32 layout: lane n = lane&15, rows j (+8 for lanes 16-31)
__device__ __forceinline__ void store_f32(v8f acc, float* D, int ldd, int m0, int n0,
                                          int lane, const float* b1, const float* b2,
                                          bool act) {
  int n  = n0 + (lane & 15);
  int mb = m0 + ((lane >> 4) << 3);
  float bv = 0.0f;
  if (b1) bv += b1[n];
  if (b2) bv += b2[n];
#pragma unroll
  for (int j = 0; j < 8; ++j) {
    float v = acc[j] + bv;
    if (act) v = tanhf(v);
    D[(size_t)(mb + j) * ldd + n] = v;
  }
}

__device__ __forceinline__ void store_bf16(v8f acc, __bf16* D, int ldd, int m0, int n0,
                                           int lane, bool act) {
  int n  = n0 + (lane & 15);
  int mb = m0 + ((lane >> 4) << 3);
#pragma unroll
  for (int j = 0; j < 8; ++j) {
    float v = acc[j];
    if (act) v = tanhf(v);
    D[(size_t)(mb + j) * ldd + n] = (__bf16)v;
  }
}

// grid-wide sense-reversing barrier (32 persistent workgroups)
__device__ __forceinline__ void gbar(unsigned* cnt, unsigned* sense, unsigned& local_sense) {
  __syncthreads();
  if (threadIdx.x == 0) {
    unsigned ls = local_sense ^ 1u;
    local_sense = ls;
    __threadfence();
    unsigned prev = atomicAdd(cnt, 1u);
    if (prev == NWG - 1u) {
      *((volatile unsigned*)cnt) = 0u;
      __threadfence();
      *((volatile unsigned*)sense) = ls;
    } else {
      while (*((volatile unsigned*)sense) != ls) { __builtin_amdgcn_s_sleep(8); }
      __threadfence();
    }
  }
  __syncthreads();
}

// One masked-softmax attention for batch row b (whole WG cooperates).
// q: [512] f32 query row; ctx: [B,512,512] f32; mask: [B,512] f32.
// Writes cvec (bf16) to catrow[0:512], copies qcat (bf16) to catrow[512:1024].
// Optionally writes normalized align probs (f32) to sc1/sc2 [512].
__device__ void attn_phase(int b, int tid,
                           const float* __restrict__ q,
                           const float* __restrict__ ctx,
                           const float* __restrict__ mask,
                           __bf16* __restrict__ catrow,
                           const __bf16* __restrict__ qcat,
                           float* sc1, float* sc2,
                           float* lds_sc, float* lds_red) {
  const float* cb = ctx + (size_t)b * 512 * 512;
  for (int l = tid; l < 512; l += NTHR) {
    const float* cr = cb + (size_t)l * 512;
    float s = 0.0f;
    for (int k = 0; k < 512; ++k) s += q[k] * cr[k];
    lds_sc[l] = (mask[(size_t)b * 512 + l] > 0.0f) ? s : -1e9f;
  }
  __syncthreads();
  // max-reduce
  lds_red[tid] = fmaxf(lds_sc[tid], lds_sc[tid + 256]);
  __syncthreads();
  for (int s = 128; s > 0; s >>= 1) {
    if (tid < s) lds_red[tid] = fmaxf(lds_red[tid], lds_red[tid + s]);
    __syncthreads();
  }
  float mx = lds_red[0];
  __syncthreads();
  // exp + sum-reduce
  float e0 = expf(lds_sc[tid] - mx);
  float e1 = expf(lds_sc[tid + 256] - mx);
  lds_sc[tid] = e0;
  lds_sc[tid + 256] = e1;
  lds_red[tid] = e0 + e1;
  __syncthreads();
  for (int s = 128; s > 0; s >>= 1) {
    if (tid < s) lds_red[tid] += lds_red[tid + s];
    __syncthreads();
  }
  float inv = 1.0f / lds_red[0];
  __syncthreads();
  lds_sc[tid] *= inv;
  lds_sc[tid + 256] *= inv;
  __syncthreads();
  if (sc1) {
    sc1[tid] = lds_sc[tid];       sc1[tid + 256] = lds_sc[tid + 256];
    sc2[tid] = lds_sc[tid];       sc2[tid + 256] = lds_sc[tid + 256];
  }
  // context vector + cat assembly
  for (int h = tid; h < 512; h += NTHR) {
    float cv = 0.0f;
    for (int l = 0; l < 512; ++l) cv += lds_sc[l] * cb[(size_t)l * 512 + h];
    catrow[h]       = (__bf16)cv;
    catrow[512 + h] = qcat[h];
  }
  __syncthreads();
}

__device__ __forceinline__ void cvt_arr(const float* __restrict__ s, __bf16* __restrict__ d,
                                        int n, int gtid) {
  for (int i = gtid; i < n; i += GSZ) d[i] = (__bf16)s[i];
}

// ---------------- persistent cooperative kernel ----------------
__global__ void __launch_bounds__(NTHR, 1)
decoder_persistent(const int* __restrict__ nt, const int* __restrict__ prevr,
                   const int* __restrict__ parr, const int* __restrict__ pidx,
                   const float* __restrict__ src_ctx, const float* __restrict__ rest_ctx,
                   const float* __restrict__ src_mask, const float* __restrict__ rest_mask,
                   const float* __restrict__ h0, const float* __restrict__ c0,
                   const float* __restrict__ emb_nt, const float* __restrict__ emb_rule,
                   const float* __restrict__ Wih, const float* __restrict__ Whh,
                   const float* __restrict__ bih, const float* __restrict__ bhh,
                   const float* __restrict__ Wins, const float* __restrict__ Wouts,
                   const float* __restrict__ Winv, const float* __restrict__ Woutv,
                   const float* __restrict__ Wal, const float* __restrict__ bal,
                   float* __restrict__ out, char* __restrict__ ws) {
  const int tid  = threadIdx.x;
  const int wg   = blockIdx.x;
  const int gtid = wg * NTHR + tid;
  const int wave = gtid >> 5;
  const int lane = tid & 31;

  unsigned* bar_cnt   = (unsigned*)(ws + OFF_BAR);
  unsigned* bar_sense = bar_cnt + 1;
  unsigned local_sense = 0;

  __bf16* wih_bf   = (__bf16*)(ws + OFF_WIH);
  __bf16* whh_bf   = (__bf16*)(ws + OFF_WHH);
  __bf16* wins_bf  = (__bf16*)(ws + OFF_WINS);
  __bf16* wouts_bf = (__bf16*)(ws + OFF_WOUTS);
  __bf16* winv_bf  = (__bf16*)(ws + OFF_WINV);
  __bf16* woutv_bf = (__bf16*)(ws + OFF_WOUTV);
  __bf16* wal_bf   = (__bf16*)(ws + OFF_WAL);
  __bf16* buf_bf   = (__bf16*)(ws + OFF_BUF);
  __bf16* x_bf     = (__bf16*)(ws + OFF_XBF);
  __bf16* cat1     = (__bf16*)(ws + OFF_CAT1);
  __bf16* cat2     = (__bf16*)(ws + OFF_CAT2);
  __bf16* cat3     = (__bf16*)(ws + OFF_CAT3);
  float*  gates    = (float*)(ws + OFF_GATES);
  float*  cst      = (float*)(ws + OFF_C);
  float*  qp       = (float*)(ws + OFF_QP);
  float*  qp2      = (float*)(ws + OFF_QP2);

  float* out_y   = out;                                    // [B,T,512]
  float* out_sc1 = out + (size_t)Bsz * Tlen * Hdim;        // [B,T,512]
  float* out_sc2 = out_sc1 + (size_t)Bsz * Tlen * Hdim;    // duplicate

  __shared__ float lds_sc[512];
  __shared__ float lds_red[256];

  // ---- init: weights -> bf16, state init ----
  cvt_arr(Wih,   wih_bf,   2048 * 2048, gtid);
  cvt_arr(Whh,   whh_bf,   2048 * 512,  gtid);
  cvt_arr(Wins,  wins_bf,  512 * 512,   gtid);
  cvt_arr(Wouts, wouts_bf, 512 * 1024,  gtid);
  cvt_arr(Winv,  winv_bf,  512 * 512,   gtid);
  cvt_arr(Woutv, woutv_bf, 512 * 1024,  gtid);
  cvt_arr(Wal,   wal_bf,   512 * 1536,  gtid);
  for (int i = gtid; i < Bsz * Hdim; i += GSZ) {
    int b = i >> 9, h = i & 511;
    cst[i] = c0[i];
    cat3[(size_t)b * 1536 + h] = (__bf16)h0[i];            // h state lives in cat3[0:512]
  }
  gbar(bar_cnt, bar_sense, local_sense);

  // ---- serial recurrence ----
  for (int t = 0; t < Tlen; ++t) {
    // assemble x = [emb_nt; emb_rule(prev); emb_rule(parent); parent_h] in bf16
    for (int idx = gtid; idx < Bsz * 2048; idx += GSZ) {
      int b = idx >> 11, k = idx & 2047;
      __bf16 v;
      if (k < 1536) {
        float f;
        if (k < 512)        f = emb_nt  [(size_t)nt  [b * Tlen + t] * 512 + k];
        else if (k < 1024)  f = emb_rule[(size_t)prevr[b * Tlen + t] * 512 + (k - 512)];
        else                f = emb_rule[(size_t)parr [b * Tlen + t] * 512 + (k - 1024)];
        v = (__bf16)f;
      } else {
        int p = pidx[b * Tlen + t];
        v = (p < t) ? buf_bf[((size_t)b * Tlen + p) * 512 + (k - 1536)] : (__bf16)0.0f;
      }
      x_bf[idx] = v;
    }
    gbar(bar_cnt, bar_sense, local_sense);

    // gates = x@Wih^T + h@Whh^T + bih + bhh : 256 tiles, one per wave
    {
      int m0 = (wave >> 7) << 4;
      int n0 = (wave & 127) << 4;
      v8f acc = {};
      acc = wmma_accum(acc, x_bf, 2048, wih_bf, 2048, m0, n0, 2048, lane);
      acc = wmma_accum(acc, cat3, 1536, whh_bf, 512,  m0, n0, 512,  lane);
      store_f32(acc, gates, 2048, m0, n0, lane, bih, bhh, false);
    }
    gbar(bar_cnt, bar_sense, local_sense);

    // LSTM elementwise -> c, h (bf16 into cat3[0:512] and history buffer)
    for (int idx = gtid; idx < Bsz * Hdim; idx += GSZ) {
      int b = idx >> 9, h = idx & 511;
      const float* g = gates + (size_t)b * 2048;
      float ig = sigf(g[h]);
      float fg = sigf(g[512 + h]);
      float gg = tanhf(g[1024 + h]);
      float og = sigf(g[1536 + h]);
      float cn = fg * cst[idx] + ig * gg;
      float hn = og * tanhf(cn);
      cst[idx] = cn;
      __bf16 hb = (__bf16)hn;
      cat3[(size_t)b * 1536 + h] = hb;
      buf_bf[((size_t)b * Tlen + t) * 512 + h] = hb;
    }
    gbar(bar_cnt, bar_sense, local_sense);

    // qp = h @ Win_s^T
    if (wave < 64) {
      int m0 = (wave >> 5) << 4, n0 = (wave & 31) << 4;
      v8f acc = {};
      acc = wmma_accum(acc, cat3, 1536, wins_bf, 512, m0, n0, 512, lane);
      store_f32(acc, qp, 512, m0, n0, lane, nullptr, nullptr, false);
    }
    gbar(bar_cnt, bar_sense, local_sense);

    // attention 1 over src_context; cat1 = [cvec1; h]
    attn_phase(wg, tid, qp + (size_t)wg * 512, src_ctx, src_mask,
               cat1 + (size_t)wg * 1024, cat3 + (size_t)wg * 1536,
               nullptr, nullptr, lds_sc, lds_red);
    gbar(bar_cnt, bar_sense, local_sense);

    // src_out = tanh(cat1 @ Wout_s^T) -> bf16 into cat3[512:1024]
    if (wave < 64) {
      int m0 = (wave >> 5) << 4, n0 = (wave & 31) << 4;
      v8f acc = {};
      acc = wmma_accum(acc, cat1, 1024, wouts_bf, 1024, m0, n0, 1024, lane);
      store_bf16(acc, cat3 + 512, 1536, m0, n0, lane, true);
    }
    gbar(bar_cnt, bar_sense, local_sense);

    // qp2 = src_out @ Win_v^T
    if (wave < 64) {
      int m0 = (wave >> 5) << 4, n0 = (wave & 31) << 4;
      v8f acc = {};
      acc = wmma_accum(acc, cat3 + 512, 1536, winv_bf, 512, m0, n0, 512, lane);
      store_f32(acc, qp2, 512, m0, n0, lane, nullptr, nullptr, false);
    }
    gbar(bar_cnt, bar_sense, local_sense);

    // attention 2 over rest_context; emits var_sc to both score outputs; cat2 = [cvec2; src_out]
    attn_phase(wg, tid, qp2 + (size_t)wg * 512, rest_ctx, rest_mask,
               cat2 + (size_t)wg * 1024, cat3 + (size_t)wg * 1536 + 512,
               out_sc1 + ((size_t)wg * Tlen + t) * 512,
               out_sc2 + ((size_t)wg * Tlen + t) * 512, lds_sc, lds_red);
    gbar(bar_cnt, bar_sense, local_sense);

    // var_out = tanh(cat2 @ Wout_v^T) -> bf16 into cat3[1024:1536]
    if (wave < 64) {
      int m0 = (wave >> 5) << 4, n0 = (wave & 31) << 4;
      v8f acc = {};
      acc = wmma_accum(acc, cat2, 1024, woutv_bf, 1024, m0, n0, 1024, lane);
      store_bf16(acc, cat3 + 1024, 1536, m0, n0, lane, true);
    }
    gbar(bar_cnt, bar_sense, local_sense);

    // a = tanh(cat3 @ Wal^T + bal) -> output[b, t, :]
    if (wave < 64) {
      int m0 = (wave >> 5) << 4, n0 = (wave & 31) << 4;
      v8f acc = {};
      acc = wmma_accum(acc, cat3, 1536, wal_bf, 1536, m0, n0, 1536, lane);
      store_f32(acc, out_y + (size_t)t * 512, Tlen * 512, m0, n0, lane, bal, nullptr, true);
    }
    gbar(bar_cnt, bar_sense, local_sense);
  }
}

extern "C" void kernel_launch(void* const* d_in, const int* in_sizes, int n_in,
                              void* d_out, int out_size, void* d_ws, size_t ws_size,
                              hipStream_t stream) {
  (void)in_sizes; (void)n_in; (void)out_size; (void)ws_size;
  // zero barrier state every call (ws is poisoned once and never re-poisoned)
  hipMemsetAsync(d_ws, 0, 256, stream);
  decoder_persistent<<<NWG, NTHR, 0, stream>>>(
      (const int*)d_in[0], (const int*)d_in[1], (const int*)d_in[2], (const int*)d_in[3],
      (const float*)d_in[4], (const float*)d_in[5], (const float*)d_in[6], (const float*)d_in[7],
      (const float*)d_in[8], (const float*)d_in[9], (const float*)d_in[10], (const float*)d_in[11],
      (const float*)d_in[12], (const float*)d_in[13], (const float*)d_in[14], (const float*)d_in[15],
      (const float*)d_in[16], (const float*)d_in[17], (const float*)d_in[18], (const float*)d_in[19],
      (const float*)d_in[20], (const float*)d_in[21],
      (float*)d_out, (char*)d_ws);
}
